// MultiHeadAttention_4174708212023
// MI455X (gfx1250) — matrix operand order repaired and checked by reading, not compiled, not run on hardware
//
#include <hip/hip_runtime.h>

#define B_  4
#define S_  2048
#define D_  512
#define H_  8
#define DK_ 64

#define SSTR (S_ + 4)   // f32 score row stride in LDS (bank-conflict padding)
#define PSTR (S_ + 8)   // f16 prob  row stride in LDS (bank-conflict padding)

typedef __attribute__((ext_vector_type(16))) _Float16 v16h;
typedef __attribute__((ext_vector_type(8)))  _Float16 v8h;
typedef __attribute__((ext_vector_type(8)))  float    v8f;

// ---------------------------------------------------------------------------
// WMMA helper: D(16x16,f32) += A(16x32,f16) * B(32x16,f16)
// ---------------------------------------------------------------------------
static __device__ __forceinline__ v8f wmma_f32_f16(v16h a, v16h b, v8f c) {
  return __builtin_amdgcn_wmma_f32_16x16x32_f16(
      /*neg_a=*/false, a, /*neg_b=*/false, b,
      /*c_mod=*/(short)0, c, /*reuse_a=*/false, /*reuse_b=*/false);
}

// A-matrix fragment 16x32 (MxK), f16 source, unit k-stride (ISA 7.12.2):
// lane l: row m = l%16, g = l/16; elements e<8 -> k=8g+e ; e>=8 -> k=16+8g+(e-8)
static __device__ __forceinline__ v16h load_frag_a_h(const _Float16* __restrict__ rowk, int g) {
  v8h lo = *(const v8h*)(rowk + 8 * g);
  v8h hi = *(const v8h*)(rowk + 16 + 8 * g);
  v16h f;
#pragma unroll
  for (int i = 0; i < 8; ++i) { f[i] = lo[i]; f[i + 8] = hi[i]; }
  return f;
}

// B-matrix fragment 32x16 (KxN), f16 source, unit k-stride:
// lane l: col n = l%16, g = l/16; elements e<8 -> k=8g+e ; e>=8 -> k=16+8g+(e-8)
static __device__ __forceinline__ v16h load_frag_b_h(const _Float16* __restrict__ colk, int g) {
  v8h lo = *(const v8h*)(colk + 8 * g);
  v8h hi = *(const v8h*)(colk + 16 + 8 * g);
  v16h f;
#pragma unroll
  for (int i = 0; i < 8; ++i) { f[i] = lo[i]; f[i + 8] = hi[i]; }
  return f;
}

// ---------------------------------------------------------------------------
// Kernel 0: streaming f32 -> f16 convert (8 elements / thread, b128 in/out).
// ---------------------------------------------------------------------------
__global__ __launch_bounds__(256)
void cvt_f32_f16_kernel(const float* __restrict__ in, _Float16* __restrict__ out) {
  const size_t i = ((size_t)blockIdx.x * 256 + threadIdx.x) * 8;
  v8h o;
#pragma unroll
  for (int e = 0; e < 8; ++e) o[e] = (_Float16)in[i + e];
  *(v8h*)(out + i) = o;
}

// ---------------------------------------------------------------------------
// Kernel 1: Out[m,n] = sum_k X[m,k]*W[n,k] + bias[n]   (x @ W^T + b)
// K = N = 512 compile-time; fully unrolled, depth-1 software pipeline:
// chunk i+1's fragments are loaded before chunk i's WMMAs are issued.
// ---------------------------------------------------------------------------
template <typename TOut>
__global__ __launch_bounds__(256)
void proj_gemm_kernel(const _Float16* __restrict__ X, const _Float16* __restrict__ W,
                      const float* __restrict__ bias, TOut* __restrict__ Out) {
  constexpr int K = D_;
  constexpr int N = D_;
  const int lane = threadIdx.x & 31;
  const int wave = threadIdx.x >> 5;
  const int g  = lane >> 4;
  const int ln = lane & 15;

  const int m0 = (blockIdx.x * 8 + wave) * 16;
  const int n0 = blockIdx.y * 64;

  v8f acc[4] = {};
  const _Float16* xrow = X + (size_t)(m0 + ln) * (size_t)K;
  const _Float16* wr0 = W + (size_t)(n0 + ln) * (size_t)K;
  const _Float16* wr1 = wr0 + (size_t)16 * K;
  const _Float16* wr2 = wr1 + (size_t)16 * K;
  const _Float16* wr3 = wr2 + (size_t)16 * K;

  v16h a  = load_frag_a_h(xrow, g);
  v16h b0 = load_frag_b_h(wr0, g);
  v16h b1 = load_frag_b_h(wr1, g);
  v16h b2 = load_frag_b_h(wr2, g);
  v16h b3 = load_frag_b_h(wr3, g);

#pragma unroll
  for (int kb = 0; kb < K; kb += 32) {
    v16h an, bn0, bn1, bn2, bn3;
    if (kb + 32 < K) {           // compile-time per unrolled iteration
      an  = load_frag_a_h(xrow + kb + 32, g);
      bn0 = load_frag_b_h(wr0 + kb + 32, g);
      bn1 = load_frag_b_h(wr1 + kb + 32, g);
      bn2 = load_frag_b_h(wr2 + kb + 32, g);
      bn3 = load_frag_b_h(wr3 + kb + 32, g);
    }
    acc[0] = wmma_f32_f16(a, b0, acc[0]);
    acc[1] = wmma_f32_f16(a, b1, acc[1]);
    acc[2] = wmma_f32_f16(a, b2, acc[2]);
    acc[3] = wmma_f32_f16(a, b3, acc[3]);
    if (kb + 32 < K) { a = an; b0 = bn0; b1 = bn1; b2 = bn2; b3 = bn3; }
  }

#pragma unroll
  for (int j = 0; j < 4; ++j) {
    const int col = n0 + 16 * j + ln;
    const float bb = bias[col];
#pragma unroll
    for (int v = 0; v < 8; ++v) {
      const int row = m0 + v + 8 * g;
      Out[(size_t)row * (size_t)N + col] = (TOut)(acc[j][v] + bb);
    }
  }
}

// ---------------------------------------------------------------------------
// Kernel 1b: V projection with TRANSPOSED f16 output: Vt[b][n][s].
// Same pipelined GEMM core; 8 consecutive `s` per lane -> one 16B v8h store.
// ---------------------------------------------------------------------------
__global__ __launch_bounds__(256)
void proj_gemm_vt_kernel(const _Float16* __restrict__ X, const _Float16* __restrict__ W,
                         const float* __restrict__ bias, _Float16* __restrict__ Vt) {
  constexpr int K = D_;
  const int lane = threadIdx.x & 31;
  const int wave = threadIdx.x >> 5;
  const int g  = lane >> 4;
  const int ln = lane & 15;

  const int m0 = (blockIdx.x * 8 + wave) * 16;
  const int n0 = blockIdx.y * 64;

  v8f acc[4] = {};
  const _Float16* xrow = X + (size_t)(m0 + ln) * (size_t)K;
  const _Float16* wr0 = W + (size_t)(n0 + ln) * (size_t)K;
  const _Float16* wr1 = wr0 + (size_t)16 * K;
  const _Float16* wr2 = wr1 + (size_t)16 * K;
  const _Float16* wr3 = wr2 + (size_t)16 * K;

  v16h a  = load_frag_a_h(xrow, g);
  v16h b0 = load_frag_b_h(wr0, g);
  v16h b1 = load_frag_b_h(wr1, g);
  v16h b2 = load_frag_b_h(wr2, g);
  v16h b3 = load_frag_b_h(wr3, g);

#pragma unroll
  for (int kb = 0; kb < K; kb += 32) {
    v16h an, bn0, bn1, bn2, bn3;
    if (kb + 32 < K) {
      an  = load_frag_a_h(xrow + kb + 32, g);
      bn0 = load_frag_b_h(wr0 + kb + 32, g);
      bn1 = load_frag_b_h(wr1 + kb + 32, g);
      bn2 = load_frag_b_h(wr2 + kb + 32, g);
      bn3 = load_frag_b_h(wr3 + kb + 32, g);
    }
    acc[0] = wmma_f32_f16(a, b0, acc[0]);
    acc[1] = wmma_f32_f16(a, b1, acc[1]);
    acc[2] = wmma_f32_f16(a, b2, acc[2]);
    acc[3] = wmma_f32_f16(a, b3, acc[3]);
    if (kb + 32 < K) { a = an; b0 = bn0; b1 = bn1; b2 = bn2; b3 = bn3; }
  }

  const int r0 = m0 + 8 * g;
  const int b  = r0 / S_;
  const int s0 = r0 % S_;          // multiple of 8 -> 16B-aligned store
#pragma unroll
  for (int j = 0; j < 4; ++j) {
    const int col = n0 + 16 * j + ln;       // = h*DK + dk
    const float bb = bias[col];
    v8h pack;
#pragma unroll
    for (int v = 0; v < 8; ++v) pack[v] = (_Float16)(acc[j][v] + bb);
    *(v8h*)&Vt[((size_t)b * D_ + col) * S_ + s0] = pack;
  }
}

// ---------------------------------------------------------------------------
// Kernel 2: one workgroup (256 thr = 8 waves, 2 waves/SIMD) owns 16 query
// rows of one (b,h).  LDS: f32 scores 16xSSTR (~128 KB) + f16 probs 16xPSTR
// (~64 KB) + reductions (1 KB) + split-K scratch (4 KB) ~= 198 KB < 320 KB.
// ---------------------------------------------------------------------------
__global__ __launch_bounds__(256)
void attn_kernel(const _Float16* __restrict__ Qh, const _Float16* __restrict__ Kh,
                 const _Float16* __restrict__ Vt, const int* __restrict__ mask,
                 float* __restrict__ probs, _Float16* __restrict__ Aout) {
  extern __shared__ float smem[];
  float*    sc     = smem;                           // [16][SSTR] f32 scores
  _Float16* p16    = (_Float16*)(smem + 16 * SSTR);  // [16][PSTR] f16 probs
  float*    red    = (float*)(p16 + 16 * PSTR);      // [256] reductions
  float*    accbuf = red + 256;                      // [4][16][16] split-K

  const int lane = threadIdx.x & 31;
  const int wave = threadIdx.x >> 5;  // 0..7
  const int g  = lane >> 4;
  const int ln = lane & 15;

  const int qt = blockIdx.x & (S_ / 16 - 1);
  const int bh = blockIdx.x / (S_ / 16);
  const int b  = bh / H_;
  const int h  = bh % H_;
  const int q0 = qt * 16;

  const size_t base = (size_t)b * S_ * D_ + (size_t)h * DK_;

  // Q fragments for this 16-row tile (DK=64 -> two 32-wide k-chunks).
  const _Float16* qrow = Qh + base + (size_t)(q0 + ln) * D_;
  const v16h qa0 = load_frag_a_h(qrow, g);
  const v16h qa1 = load_frag_a_h(qrow + 32, g);

  const int* mbase = mask + (size_t)b * S_ * S_;

  // ---- phase 1: scores = (Q K^T)/8, masked, into LDS ----------------------
  // Depth-1 pipeline: next tile's K-fragments load across this tile's WMMAs
  // and mask/LDS epilogue.
  {
    const _Float16* krow = Kh + base + (size_t)(wave * 16 + ln) * D_;
    v16h kb0 = load_frag_b_h(krow, g);
    v16h kb1 = load_frag_b_h(krow + 32, g);
    for (int ct = wave; ct < S_ / 16; ct += 8) {
      const bool more = (ct + 8) < (S_ / 16);
      v16h nb0, nb1;
      if (more) {
        const _Float16* nk = Kh + base + (size_t)((ct + 8) * 16 + ln) * D_;
        nb0 = load_frag_b_h(nk, g);
        nb1 = load_frag_b_h(nk + 32, g);
      }
      v8f acc = {};
      acc = wmma_f32_f16(qa0, kb0, acc);
      acc = wmma_f32_f16(qa1, kb1, acc);
      const int n0 = ct * 16;
#pragma unroll
      for (int v = 0; v < 8; ++v) {
        const int m   = v + 8 * g;
        const int col = n0 + ln;
        float s = acc[v] * 0.125f;     // 1/sqrt(64)
        const int mk = mbase[(size_t)(q0 + m) * S_ + col];
        s = (mk == 0) ? -1e10f : s;
        sc[m * SSTR + col] = s;
      }
      if (more) { kb0 = nb0; kb1 = nb1; }
    }
  }
  __syncthreads();

  // ---- phase 2: row softmax (16 thr/row); one HBM write of P + f16 copy ---
  const int row = threadIdx.x >> 4;  // 16 rows, 16 threads/row
  const int seg = threadIdx.x & 15;
  float* r = sc + row * SSTR;

  float mx = -3.4e38f;
  for (int c = seg; c < S_; c += 16) mx = fmaxf(mx, r[c]);
  red[threadIdx.x] = mx;
  __syncthreads();
  float rowmax = red[row * 16];
#pragma unroll
  for (int i = 1; i < 16; ++i) rowmax = fmaxf(rowmax, red[row * 16 + i]);
  __syncthreads();

  float sum = 0.f;
  for (int c = seg; c < S_; c += 16) {
    const float e = __expf(r[c] - rowmax);
    r[c] = e;
    sum += e;
  }
  red[threadIdx.x] = sum;
  __syncthreads();
  float rowsum = 0.f;
#pragma unroll
  for (int i = 0; i < 16; ++i) rowsum += red[row * 16 + i];
  const float inv = 1.0f / rowsum;

  float*     prow = probs + ((size_t)bh * S_ + (q0 + row)) * S_;
  _Float16*  hrow = p16 + row * PSTR;
  for (int c = seg; c < S_; c += 16) {
    const float p = r[c] * inv;
    prow[c] = p;                 // the one-and-only HBM write of attn_probs
    hrow[c] = (_Float16)p;       // f16 copy for the PV matmul
  }
  __syncthreads();

  // ---- phase 3: A = P @ V, split-K across wave pairs ----------------------
  const int nt   = wave & 3;
  const int half = wave >> 2;
  const int n0   = nt * 16;
  const int k0   = half * (S_ / 2);

  const _Float16* vrow = Vt + ((size_t)b * D_ + h * DK_ + n0 + ln) * S_;
  const _Float16* pr0  = p16 + ln * PSTR;
  v8f acc = {};
  v16h vb = load_frag_b_h(vrow + k0, g);
  for (int kc = k0; kc < k0 + S_ / 2; kc += 32) {
    const bool more = (kc + 32) < (k0 + S_ / 2);
    v16h nvb;
    if (more) nvb = load_frag_b_h(vrow + kc + 32, g);
    const _Float16* pr = pr0 + kc;
    v8h lo = *(const v8h*)(pr + 8 * g);
    v8h hi = *(const v8h*)(pr + 16 + 8 * g);
    v16h pa;
#pragma unroll
    for (int i = 0; i < 8; ++i) { pa[i] = lo[i]; pa[i + 8] = hi[i]; }
    acc = wmma_f32_f16(pa, vb, acc);
    if (more) vb = nvb;
  }

  // combine the two half-K partial accumulators through LDS
  float* abuf = accbuf + nt * 256;   // [16][16] for this n-tile
  if (half == 1) {
#pragma unroll
    for (int v = 0; v < 8; ++v) abuf[(v + 8 * g) * 16 + ln] = acc[v];
  }
  __syncthreads();
  if (half == 0) {
#pragma unroll
    for (int v = 0; v < 8; ++v) {
      const int m = v + 8 * g;
      const float a = acc[v] + abuf[m * 16 + ln];
      Aout[base + (size_t)(q0 + m) * D_ + n0 + ln] = (_Float16)a;
    }
  }
}

// ---------------------------------------------------------------------------
// Host launcher
// ---------------------------------------------------------------------------
extern "C" void kernel_launch(void* const* d_in, const int* in_sizes, int n_in,
                              void* d_out, int out_size, void* d_ws, size_t ws_size,
                              hipStream_t stream) {
  (void)in_sizes; (void)n_in; (void)out_size; (void)ws_size;

  const float* query = (const float*)d_in[0];
  const float* key   = (const float*)d_in[1];
  const float* value = (const float*)d_in[2];
  const int*   mask  = (const int*)d_in[3];
  const float* Wq = (const float*)d_in[4];  const float* bq = (const float*)d_in[5];
  const float* Wk = (const float*)d_in[6];  const float* bk = (const float*)d_in[7];
  const float* Wv = (const float*)d_in[8];  const float* bv = (const float*)d_in[9];
  const float* Wo = (const float*)d_in[10]; const float* bo = (const float*)d_in[11];

  float* out   = (float*)d_out;                  // (B,S,D)
  float* probs = out + (size_t)B_ * S_ * D_;     // (B,H,S,S)

  const size_t elems = (size_t)B_ * S_ * D_;     // 4 Mi
  _Float16* Qh = (_Float16*)d_ws;                // row-major  (B,S,D) f16
  _Float16* Kh = Qh + elems;                     // row-major  (B,S,D) f16
  _Float16* Vt = Kh + elems;                     // TRANSPOSED (B,D,S) f16
  _Float16* Xh = Vt + elems;                     // staging X f16; later Ah
  _Float16* Wh = Xh + elems;                     // staging W f16 (512 KB)
  _Float16* Ah = Xh;                             // recycled after last proj

  const int M = B_ * S_;                         // 8192
  dim3 pgrid(M / (16 * 8), D_ / 64);             // (64, 8)
  const int xcvt = (int)(elems / 8 / 256);       // 2048 blocks
  const int wcvt = (int)((size_t)D_ * D_ / 8 / 256);  // 128 blocks

  // Q projection
  cvt_f32_f16_kernel<<<xcvt, 256, 0, stream>>>(query, Xh);
  cvt_f32_f16_kernel<<<wcvt, 256, 0, stream>>>(Wq, Wh);
  proj_gemm_kernel<_Float16><<<pgrid, 256, 0, stream>>>(Xh, Wh, bq, Qh);
  // K projection
  cvt_f32_f16_kernel<<<xcvt, 256, 0, stream>>>(key, Xh);
  cvt_f32_f16_kernel<<<wcvt, 256, 0, stream>>>(Wk, Wh);
  proj_gemm_kernel<_Float16><<<pgrid, 256, 0, stream>>>(Xh, Wh, bk, Kh);
  // V projection (transposed output)
  cvt_f32_f16_kernel<<<xcvt, 256, 0, stream>>>(value, Xh);
  cvt_f32_f16_kernel<<<wcvt, 256, 0, stream>>>(Wv, Wh);
  proj_gemm_vt_kernel<<<pgrid, 256, 0, stream>>>(Xh, Wh, bv, Vt);

  // Attention (Ah recycles the Xh buffer; stream-ordered so Xh is dead)
  const size_t smem_bytes = (size_t)16 * SSTR * sizeof(float)
                          + (size_t)16 * PSTR * sizeof(_Float16)
                          + (256 + 4 * 256) * sizeof(float);   // ~198 KB
  attn_kernel<<<B_ * H_ * (S_ / 16), 256, smem_bytes, stream>>>(Qh, Kh, Vt, mask, probs, Ah);

  // Output projection
  cvt_f32_f16_kernel<<<wcvt, 256, 0, stream>>>(Wo, Wh);
  proj_gemm_kernel<float><<<pgrid, 256, 0, stream>>>(Ah, Wh, bo, out);
}
